// SpatialMultiHeadCrossAttention_6803228197052
// MI455X (gfx1250) — matrix operand order repaired and checked by reading, not compiled, not run on hardware
//
#include <hip/hip_runtime.h>

// ---------------------------------------------------------------------------
// SpatialMultiHeadCrossAttention for MI455X (gfx1250), wave32 + WMMA f16.
// B=2, N=512, T=12, E=128, H=8, D=16.
// ---------------------------------------------------------------------------

typedef __attribute__((ext_vector_type(16))) _Float16 v16h;
typedef __attribute__((ext_vector_type(8)))  float    v8f;

#define BB 2
#define NN 512
#define TT 12
#define EE 128
#define HH 8
#define DD 16
#define ROWS (BB * NN * TT)                  // 12288
#define HEAD_ELEMS (BB * HH * TT * NN * DD)  // 1572864 (== B*N*T*E)

static __device__ __forceinline__ v8f wmma32(v16h a, v16h b, v8f c) {
    // D = A(16x32 f16) * B(32x16 f16) + C(16x16 f32)
    return __builtin_amdgcn_wmma_f32_16x16x32_f16(
        /*neg_a=*/false, a, /*neg_b=*/false, b,
        /*c_mod=*/(short)0, c, /*reuse_a=*/false, /*reuse_b=*/false);
}

// A-operand (16x32 f16) half index -> K index:
//   lanes 0-15 (g=0): halves 0..7 -> K 0..7,  halves 8..15 -> K 16..23
//   lanes16-31 (g=1): halves 0..7 -> K 8..15, halves 8..15 -> K 24..31
// So per lane: two contiguous 8-half chunks at row+8g and row+16+8g.

// ---------------------------------------------------------------------------
// Stage 1: input projections  y = x @ W  -> f16 heads [B,H,T,N,D]
//   mid 0..5 = Qf,Kf,Vf,Qs,Ks,Vs ; mid==3 additionally emits Qf_from_speed(id 6)
// ---------------------------------------------------------------------------
__global__ __launch_bounds__(256) void proj_in_kernel(
    const float* __restrict__ fq, const float* __restrict__ fk, const float* __restrict__ fv,
    const float* __restrict__ sq, const float* __restrict__ sk, const float* __restrict__ sv,
    const float* __restrict__ WfQ, const float* __restrict__ WfK, const float* __restrict__ WfV,
    const float* __restrict__ WsQ, const float* __restrict__ WsK, const float* __restrict__ WsV,
    const float* __restrict__ Kj, const float* __restrict__ Vfp,
    _Float16* __restrict__ heads)
{
    __shared__ _Float16 Xs[64 * 128];   // 16 KB: X block rows (f16)
    __shared__ _Float16 Wt[128 * 128];  // 32 KB: W transposed [e_out][k]

    const int tid = threadIdx.x;
    const int mid = blockIdx.y;
    const int rowbase = blockIdx.x * 64;

    const float* X;
    const float* W;
    if      (mid == 0) { X = fq; W = WfQ; }
    else if (mid == 1) { X = fk; W = WfK; }
    else if (mid == 2) { X = fv; W = WfV; }
    else if (mid == 3) { X = sq; W = WsQ; }
    else if (mid == 4) { X = sk; W = WsK; }
    else               { X = sv; W = WsV; }

    // gfx1250 prefetch path (global_prefetch_b8) for the X row block
    if (tid < 64) {
        __builtin_prefetch(&X[(size_t)(rowbase + tid) * 128], 0, 1);
    }

    for (int i = tid; i < 128 * 128; i += 256) {
        int e = i >> 7, k = i & 127;
        Wt[e * 128 + k] = (_Float16)W[k * 128 + e];
    }
    for (int i = tid; i < 64 * 128; i += 256) {
        int r = i >> 7, k = i & 127;
        Xs[i] = (_Float16)X[(size_t)(rowbase + r) * 128 + k];
    }
    __syncthreads();

    const int wave = tid >> 5, lane = tid & 31;
    const int g = lane >> 4, m = lane & 15;
    const int rt  = wave >> 1;        // row tile 0..3
    const int ct0 = (wave & 1) * 4;   // col tiles 0..3 or 4..7
    const _Float16* xrow = &Xs[(rt * 16 + m) * 128];

    // A fragments are column-tile invariant: load once for all 4 ct tiles.
    v16h afrag[4];
#pragma unroll
    for (int ks = 0; ks < 4; ++ks) {
        const _Float16* ap = xrow + ks * 32 + 8 * g;
#pragma unroll
        for (int i = 0; i < 8; ++i) { afrag[ks][i] = ap[i]; afrag[ks][8 + i] = ap[16 + i]; }
    }

    for (int ci = 0; ci < 4; ++ci) {
        const int ct = ct0 + ci;
        // batch all B fragments so the ds_loads issue before the WMMA chain
        v16h bfrag[4];
#pragma unroll
        for (int ks = 0; ks < 4; ++ks) {
            const _Float16* bp = &Wt[(ct * 16 + m) * 128 + ks * 32 + 8 * g];
#pragma unroll
            for (int i = 0; i < 8; ++i) { bfrag[ks][i] = bp[i]; bfrag[ks][8 + i] = bp[16 + i]; }
        }
        v8f c = {};
#pragma unroll
        for (int ks = 0; ks < 4; ++ks) c = wmma32(afrag[ks], bfrag[ks], c);

        // scatter into [B,H,T,N,D]
        const int e = ct * 16 + m;
        const int h = e >> 4, d = e & 15;
#pragma unroll
        for (int v = 0; v < 8; ++v) {
            const int r  = rowbase + rt * 16 + v + 8 * g;   // r = (b*N + n)*T + t
            const int b_ = r / (NN * TT);
            const int n  = (r / TT) % NN;
            const int t  = r % TT;
            const size_t oidx = ((size_t)(((b_ * HH + h) * TT + t)) * NN + n) * DD + d;
            const float val = c[v];
            heads[(size_t)mid * HEAD_ELEMS + oidx] = (_Float16)val;
            if (mid == 3) {
                const float qfs =
                    Kj[n] * (val - __fdividef(val * val, Vfp[n] + 1e-5f));
                heads[(size_t)6 * HEAD_ELEMS + oidx] = (_Float16)qfs;
            }
        }
    }
}

// ---------------------------------------------------------------------------
// Stage 2: flash attention over N=512 per (b,h,t,type).
//   type 0 ff: A=Qf  B=Kf  V=Vf
//   type 1 fs: A=Kf  B=Qfs V=Vf
//   type 2 sf: A=Ks  B=Qf  V=Vs
//   type 3 ss: A=Qs  B=Ks  V=Vs
// ctx output layout: [type][B,N,T,E] f16 (proj_out input order).
// ---------------------------------------------------------------------------
__global__ __launch_bounds__(256) void attn_kernel(
    const _Float16* __restrict__ heads, _Float16* __restrict__ ctx)
{
    __shared__ _Float16 Kls[NN * DD];     // 16 KB: key rows [n][d]
    __shared__ _Float16 Vt [DD * NN];     // 16 KB: V transposed [d][n]
    __shared__ _Float16 Pls[8 * 16 * 32]; //  8 KB: per-wave P staging

    const int tid  = threadIdx.x;
    const int qblk = blockIdx.x;   // 0..3  (128 query rows each)
    const int bht  = blockIdx.y;   // 0..191 : ((b*H+h)*T + t)
    const int type = blockIdx.z;   // 0..3
    const int b_ = bht / (HH * TT);
    const int h  = (bht / TT) % HH;
    const int t  = bht % TT;

    const int AidT[4] = {0, 1, 4, 3};
    const int BidT[4] = {1, 6, 0, 4};
    const int VidT[4] = {2, 2, 5, 5};
    const size_t slice = (size_t)bht * (NN * DD);

    const _Float16* Asl = heads + (size_t)AidT[type] * HEAD_ELEMS + slice;
    const _Float16* Bsl = heads + (size_t)BidT[type] * HEAD_ELEMS + slice;
    const _Float16* Vsl = heads + (size_t)VidT[type] * HEAD_ELEMS + slice;

    // K rows, vectorized 16B copies
    {
        const uint4* src = (const uint4*)Bsl;   // 8192 halves = 1024 uint4
        uint4* dst = (uint4*)Kls;
        for (int i = tid; i < 1024; i += 256) dst[i] = src[i];
    }
    // V transposed
    for (int i = tid; i < NN * DD; i += 256) {
        const int n = i >> 4, d = i & 15;
        Vt[d * NN + n] = Vsl[i];
    }
    __syncthreads();

    const int wave = tid >> 5, lane = tid & 31;
    const int g = lane >> 4, m = lane & 15;
    const int qbase = qblk * 128 + wave * 16;

    // Q fragment: A-operand, K-dim = D = 16 padded to 32; fold scale=0.25.
    v16h aq;
    {
        const _Float16* qp = Asl + (size_t)(qbase + m) * DD + 8 * g;
#pragma unroll
        for (int i = 0; i < 8; ++i) {
            aq[i]     = (_Float16)(qp[i] * (_Float16)0.25f);
            aq[8 + i] = (_Float16)0.0f;
        }
    }

    float mrow[8], lrow[8];
    v8f o = {};
#pragma unroll
    for (int v = 0; v < 8; ++v) { mrow[v] = -1e30f; lrow[v] = 0.0f; }

    _Float16* Pw = &Pls[wave * 16 * 32];

    for (int jc = 0; jc < 16; ++jc) {     // 32 keys per chunk
        const int jb = jc * 32;
        v16h bk0, bk1;
        {
            const _Float16* k0 = &Kls[(jb + m) * DD];
            const _Float16* k1 = &Kls[(jb + 16 + m) * DD];
#pragma unroll
            for (int i = 0; i < 16; ++i) {
                const _Float16 kv0 = k0[(i & 7) + 8 * g], kv1 = k1[(i & 7) + 8 * g];
                bk0[i] = (i >= 8) ? (_Float16)0.0f : kv0;   // K rows 16..31 are pad = 0
                bk1[i] = (i >= 8) ? (_Float16)0.0f : kv1;
            }
        }
        const v8f zacc = {};
        v8f s0 = wmma32(aq, bk0, zacc);
        v8f s1 = wmma32(aq, bk1, zacc);

        // online softmax; a row lives across one 16-lane half (lanes share g)
#pragma unroll
        for (int v = 0; v < 8; ++v) {
            const float t0 = s0[v], t1 = s1[v];
            float cmax = fmaxf(t0, t1);
            cmax = fmaxf(cmax, __shfl_xor(cmax, 1));
            cmax = fmaxf(cmax, __shfl_xor(cmax, 2));
            cmax = fmaxf(cmax, __shfl_xor(cmax, 4));
            cmax = fmaxf(cmax, __shfl_xor(cmax, 8));
            const float mn = fmaxf(mrow[v], cmax);
            const float alpha = __expf(mrow[v] - mn);
            const float p0 = __expf(t0 - mn);
            const float p1 = __expf(t1 - mn);
            float rs = p0 + p1;
            rs += __shfl_xor(rs, 1);
            rs += __shfl_xor(rs, 2);
            rs += __shfl_xor(rs, 4);
            rs += __shfl_xor(rs, 8);
            lrow[v] = lrow[v] * alpha + rs;
            mrow[v] = mn;
            o[v] = o[v] * alpha;
            const int row = v + 8 * g;            // C-layout row of this lane
            Pw[row * 32 + m]      = (_Float16)p0; // col = m      (tile 0)
            Pw[row * 32 + 16 + m] = (_Float16)p1; // col = 16 + m (tile 1)
        }

        // P (C-layout in LDS, row-major 16x32) -> A-operand; V chunk -> B.
        v16h ap, bv;
        {
            const _Float16* pr = &Pw[m * 32 + 8 * g];
#pragma unroll
            for (int i = 0; i < 8; ++i) { ap[i] = pr[i]; ap[8 + i] = pr[16 + i]; }
            const _Float16* vp = &Vt[m * NN + jb + 8 * g];
#pragma unroll
            for (int i = 0; i < 8; ++i) { bv[i] = vp[i]; bv[8 + i] = vp[16 + i]; }
        }
        o = wmma32(ap, bv, o);
    }

    // normalize and write ctx[type] in [B,N,T,E] order (e = h*16 + d, d = m)
#pragma unroll
    for (int v = 0; v < 8; ++v) {
        const int n = qbase + v + 8 * g;
        const float val = __fdividef(o[v], lrow[v]);
        const size_t oidx = (size_t)type * ((size_t)ROWS * EE)
                          + ((size_t)(b_ * NN + n) * TT + t) * (size_t)EE
                          + h * DD + m;
        ctx[oidx] = (_Float16)val;
    }
}

// ---------------------------------------------------------------------------
// Stage 3: out[type] = ctx[type] @ Wout + bout   (fp32 stores to d_out)
// ---------------------------------------------------------------------------
__global__ __launch_bounds__(256) void proj_out_kernel(
    const _Float16* __restrict__ ctx, const float* __restrict__ Wout,
    const float* __restrict__ bout, float* __restrict__ out)
{
    __shared__ _Float16 Xs[64 * 128];
    __shared__ _Float16 Wt[128 * 128];

    const int tid = threadIdx.x;
    const int type = blockIdx.y;
    const int rowbase = blockIdx.x * 64;
    const _Float16* X = ctx + (size_t)type * ((size_t)ROWS * EE);
    float* O = out + (size_t)type * ((size_t)ROWS * EE);

    // gfx1250 prefetch path for the ctx row block
    if (tid < 64) {
        __builtin_prefetch(&X[(size_t)(rowbase + tid) * 128], 0, 1);
    }

    for (int i = tid; i < 128 * 128; i += 256) {
        int e = i >> 7, k = i & 127;
        Wt[e * 128 + k] = (_Float16)Wout[k * 128 + e];
    }
    {
        const uint4* src = (const uint4*)(X + (size_t)rowbase * 128); // 1024 uint4
        uint4* dst = (uint4*)Xs;
        for (int i = tid; i < 1024; i += 256) dst[i] = src[i];
    }
    __syncthreads();

    const int wave = tid >> 5, lane = tid & 31;
    const int g = lane >> 4, m = lane & 15;
    const int rt = wave >> 1;
    const int ct0 = (wave & 1) * 4;
    const _Float16* xrow = &Xs[(rt * 16 + m) * 128];

    v16h afrag[4];
#pragma unroll
    for (int ks = 0; ks < 4; ++ks) {
        const _Float16* ap = xrow + ks * 32 + 8 * g;
#pragma unroll
        for (int i = 0; i < 8; ++i) { afrag[ks][i] = ap[i]; afrag[ks][8 + i] = ap[16 + i]; }
    }

    for (int ci = 0; ci < 4; ++ci) {
        const int ct = ct0 + ci;
        v16h bfrag[4];
#pragma unroll
        for (int ks = 0; ks < 4; ++ks) {
            const _Float16* bp = &Wt[(ct * 16 + m) * 128 + ks * 32 + 8 * g];
#pragma unroll
            for (int i = 0; i < 8; ++i) { bfrag[ks][i] = bp[i]; bfrag[ks][8 + i] = bp[16 + i]; }
        }
        v8f c = {};
#pragma unroll
        for (int ks = 0; ks < 4; ++ks) c = wmma32(afrag[ks], bfrag[ks], c);

        const int e = ct * 16 + m;
        const float be = bout[e];
#pragma unroll
        for (int v = 0; v < 8; ++v) {
            const int r = rowbase + rt * 16 + v + 8 * g;
            O[(size_t)r * EE + e] = c[v] + be;
        }
    }
}

// ---------------------------------------------------------------------------
extern "C" void kernel_launch(void* const* d_in, const int* in_sizes, int n_in,
                              void* d_out, int out_size, void* d_ws, size_t ws_size,
                              hipStream_t stream)
{
    (void)in_sizes; (void)n_in; (void)out_size; (void)ws_size;

    const float* fq  = (const float*)d_in[0];
    const float* fk  = (const float*)d_in[1];
    const float* fv  = (const float*)d_in[2];
    const float* sq  = (const float*)d_in[3];
    const float* sk  = (const float*)d_in[4];
    const float* sv  = (const float*)d_in[5];
    const float* WfQ = (const float*)d_in[6];
    const float* WfK = (const float*)d_in[7];
    const float* WfV = (const float*)d_in[8];
    const float* WsQ = (const float*)d_in[9];
    const float* WsK = (const float*)d_in[10];
    const float* WsV = (const float*)d_in[11];
    const float* Kj  = (const float*)d_in[12];
    const float* Vfp = (const float*)d_in[13];
    const float* Wout = (const float*)d_in[14];
    const float* bout = (const float*)d_in[15];

    _Float16* heads = (_Float16*)d_ws;                       // 7 * HEAD_ELEMS f16
    _Float16* ctx   = heads + (size_t)7 * HEAD_ELEMS;        // 4 * ROWS*EE   f16

    dim3 g1(ROWS / 64, 6);
    proj_in_kernel<<<g1, 256, 0, stream>>>(fq, fk, fv, sq, sk, sv,
                                           WfQ, WfK, WfV, WsQ, WsK, WsV,
                                           Kj, Vfp, heads);

    dim3 g2(NN / 128, BB * HH * TT, 4);  // (4, 192, 4)
    attn_kernel<<<g2, 256, 0, stream>>>(heads, ctx);

    dim3 g3(ROWS / 64, 4);
    proj_out_kernel<<<g3, 256, 0, stream>>>(ctx, Wout, bout, (float*)d_out);
}
